// TextEncoder_86165633892595
// MI455X (gfx1250) — matrix operand order repaired and checked by reading, not compiled, not run on hardware
//
#include <hip/hip_runtime.h>
#include <math.h>

typedef __bf16 bf16_t;
typedef __attribute__((ext_vector_type(16))) __bf16 v16bf;
typedef __attribute__((ext_vector_type(8)))  __bf16 v8bf;
typedef __attribute__((ext_vector_type(4)))  __bf16 v4bf;
typedef __attribute__((ext_vector_type(8)))  float  v8f;

namespace {
constexpr int CTXL  = 77;
constexpr int WID   = 512;
constexpr int NLAY  = 12;
constexpr int EMB   = 1024;
constexpr int NSEQ  = 128;   // N
constexpr int NP    = 8;     // P
constexpr int LPRM  = 16;    // LP
constexpr int EOTID = 49407;
constexpr int BSEQ  = NP * NSEQ;     // 1024 sequences
constexpr int NTOK  = BSEQ * CTXL;   // 78848 tokens
}

__device__ __forceinline__ float waveRedAdd(float v) {
  for (int o = 16; o > 0; o >>= 1) v += __shfl_xor(v, o, 32);
  return v;
}

// ---------------- fp32 -> bf16 conversion ----------------
__global__ __launch_bounds__(256) void f2bf_kernel(const float* __restrict__ src,
                                                   bf16_t* __restrict__ dst, size_t n) {
  size_t i = (size_t)blockIdx.x * blockDim.x + threadIdx.x;
  if (i < n) dst[i] = (bf16_t)src[i];
}

// transpose-convert text_projection (512 x 1024) -> tpT (1024 x 512)
__global__ __launch_bounds__(256) void f2bf_t_kernel(const float* __restrict__ src,
                                                     bf16_t* __restrict__ dst) {
  int i = blockIdx.x * blockDim.x + threadIdx.x;
  if (i < WID * EMB) {
    int k = i >> 10, e = i & 1023;        // src[k, e]
    dst[(size_t)e * WID + k] = (bf16_t)src[i];
  }
}

// ---------------- pos_y = first EOT position ----------------
__global__ void posy_kernel(const int* __restrict__ text, int* __restrict__ posy) {
  int n = blockIdx.x * blockDim.x + threadIdx.x;
  if (n < NSEQ) {
    int p = 0;
    for (int t = 0; t < CTXL; ++t)
      if (text[n * CTXL + t] == EOTID) { p = t; break; }
    posy[n] = p;
  }
}

// ---------------- embedding + prompt splice (wave per token) ----------------
__global__ __launch_bounds__(256) void embed_kernel(const int* __restrict__ text,
                                                    const float* __restrict__ prompt,
                                                    const float* __restrict__ tok_emb,
                                                    const float* __restrict__ pos_emb,
                                                    const int* __restrict__ posy,
                                                    float* __restrict__ h) {
  int tok = blockIdx.x * 8 + (threadIdx.x >> 5);
  if (tok >= NTOK) return;
  int b = tok / CTXL, t = tok - b * CTXL;
  int p = b / NSEQ, n = b - p * NSEQ;
  int py = posy[n];
  const float* src;
  if (t >= py && t < py + LPRM) {
    src = prompt + ((size_t)p * LPRM + (t - py)) * WID;
  } else {
    int s = (t < py) ? t : (t - LPRM + 1);
    s = min(max(s, 0), CTXL - 1);
    src = tok_emb + (size_t)text[n * CTXL + s] * WID;
  }
  const float* pe = pos_emb + (size_t)t * WID;
  float* out = h + (size_t)tok * WID;
  int lane = threadIdx.x & 31;
  for (int j = 0; j < 4; ++j) {
    int idx = j * 32 + lane;                     // float4 index, 4*32*4 = 512
    float4 a = ((const float4*)src)[idx];
    float4 c = ((const float4*)pe)[idx];
    float4 o = make_float4(a.x + c.x, a.y + c.y, a.z + c.z, a.w + c.w);
    ((float4*)out)[idx] = o;
  }
}

// ---------------- LayerNorm row -> bf16 (wave per row of 512) ----------------
__device__ __forceinline__ void ln_row(const float* __restrict__ x,
                                       const float* __restrict__ g,
                                       const float* __restrict__ bt,
                                       bf16_t* __restrict__ y, int lane) {
  float4 xv[4];
  float s = 0.f, sq = 0.f;
  for (int j = 0; j < 4; ++j) {
    xv[j] = ((const float4*)x)[j * 32 + lane];
    s  += xv[j].x + xv[j].y + xv[j].z + xv[j].w;
    sq += xv[j].x * xv[j].x + xv[j].y * xv[j].y + xv[j].z * xv[j].z + xv[j].w * xv[j].w;
  }
  s = waveRedAdd(s); sq = waveRedAdd(sq);
  float mean = s * (1.f / 512.f);
  float var  = sq * (1.f / 512.f) - mean * mean;
  float rs   = rsqrtf(var + 1e-5f);
  for (int j = 0; j < 4; ++j) {
    int idx = j * 32 + lane;
    float4 gv = ((const float4*)g)[idx];
    float4 bv = ((const float4*)bt)[idx];
    v4bf o;
    o[0] = (bf16_t)((xv[j].x - mean) * rs * gv.x + bv.x);
    o[1] = (bf16_t)((xv[j].y - mean) * rs * gv.y + bv.y);
    o[2] = (bf16_t)((xv[j].z - mean) * rs * gv.z + bv.z);
    o[3] = (bf16_t)((xv[j].w - mean) * rs * gv.w + bv.w);
    ((v4bf*)y)[idx] = o;
  }
}

__global__ __launch_bounds__(256) void ln_kernel(const float* __restrict__ X,
                                                 const float* __restrict__ g,
                                                 const float* __restrict__ bt,
                                                 bf16_t* __restrict__ Y, int rows) {
  int r = blockIdx.x * 8 + (threadIdx.x >> 5);
  if (r >= rows) return;
  ln_row(X + (size_t)r * WID, g, bt, Y + (size_t)r * WID, threadIdx.x & 31);
}

// ---------------- WMMA NT GEMM: C[M,N] = A[M,K] * W[N,K]^T + bias ----------------
// Wave computes a 32(M) x 64(N) tile: 2x4 accumulators; A frags reused across 4
// N-tiles, B frags reused across 2 M-tiles -> 1.5 b128 loads per WMMA.
// MODE 0: store bf16   1: gelu -> bf16   2: fp32 Out += v   3: store fp32
template <int MODE>
__global__ __launch_bounds__(256) void gemm_nt_wmma(const bf16_t* __restrict__ A,
                                                    const bf16_t* __restrict__ Wt,
                                                    const float* __restrict__ bias,
                                                    void* __restrict__ Out,
                                                    int M, int N, int K) {
  int gwave = blockIdx.x * 8 + (threadIdx.x >> 5);
  int ntn = N >> 6;                       // 64-wide column tiles
  int tm = gwave / ntn, tn = gwave - tm * ntn;
  if (tm >= (M >> 5)) return;             // 32-tall row tiles
  int lane = threadIdx.x & 31;
  int lo = lane & 15, hi = lane >> 4;
  const bf16_t* arow0 = A + (size_t)(tm * 32 + lo) * K;
  const bf16_t* arow1 = arow0 + (size_t)16 * K;
  const bf16_t* wrow  = Wt + (size_t)(tn * 64 + lo) * K;
  const size_t wstep  = (size_t)16 * K;
  int ca = hi * 8;     // A K-chunk offsets: {ca, ca+16}
  int cb = hi * 8;     // B K-chunk offsets: {cb, cb+16}

  // warm WGP$ for the streamed rows (global_prefetch_b8)
  __builtin_prefetch(arow0, 0, 1);
  __builtin_prefetch(arow1, 0, 1);
  __builtin_prefetch(wrow, 0, 1);
  __builtin_prefetch(wrow + 2 * wstep, 0, 1);

  const v8f z = {0.f, 0.f, 0.f, 0.f, 0.f, 0.f, 0.f, 0.f};
  v8f acc[2][4];
#pragma unroll
  for (int i = 0; i < 2; ++i)
#pragma unroll
    for (int j = 0; j < 4; ++j) acc[i][j] = z;

  union U { v16bf v; v8bf h[2]; };
  for (int k0 = 0; k0 < K; k0 += 32) {
    U a0, a1;
    a0.h[0] = *(const v8bf*)(arow0 + k0 + ca);
    a0.h[1] = *(const v8bf*)(arow0 + k0 + ca + 16);
    a1.h[0] = *(const v8bf*)(arow1 + k0 + ca);
    a1.h[1] = *(const v8bf*)(arow1 + k0 + ca + 16);
#pragma unroll
    for (int j = 0; j < 4; ++j) {
      U b;
      const bf16_t* wp = wrow + (size_t)j * wstep + k0 + cb;
      b.h[0] = *(const v8bf*)(wp);
      b.h[1] = *(const v8bf*)(wp + 16);
      acc[0][j] = __builtin_amdgcn_wmma_f32_16x16x32_bf16(false, a0.v, false, b.v,
                                                          (short)0, acc[0][j], false, false);
      acc[1][j] = __builtin_amdgcn_wmma_f32_16x16x32_bf16(false, a1.v, false, b.v,
                                                          (short)0, acc[1][j], false, false);
    }
  }

#pragma unroll
  for (int j = 0; j < 4; ++j) {
    int n = tn * 64 + j * 16 + lo;
    float bv = bias ? bias[n] : 0.f;
#pragma unroll
    for (int i = 0; i < 2; ++i) {
#pragma unroll
      for (int r = 0; r < 8; ++r) {
        int m = tm * 32 + i * 16 + r + hi * 8;
        float v = acc[i][j][r] + bv;
        size_t idx = (size_t)m * N + n;
        if (MODE == 0) {
          ((bf16_t*)Out)[idx] = (bf16_t)v;
        } else if (MODE == 1) {
          float gl = v / (1.f + __expf(-1.702f * v));
          ((bf16_t*)Out)[idx] = (bf16_t)gl;
        } else if (MODE == 2) {
          float* o = (float*)Out + idx;
          *o = *o + v;
        } else {
          ((float*)Out)[idx] = v;
        }
      }
    }
  }
}

// ---------------- attention: block per (batch, head), wave per query row ----------------
__global__ __launch_bounds__(256) void attn_kernel(const bf16_t* __restrict__ qkv,
                                                   bf16_t* __restrict__ out) {
  int bh = blockIdx.x;
  int b = bh >> 3, hd = bh & 7;
  __shared__ bf16_t ks[CTXL * 64];
  __shared__ bf16_t vs[CTXL * 64];
  const bf16_t* base = qkv + (size_t)b * CTXL * 1536;
  for (int i = threadIdx.x; i < CTXL * 64; i += blockDim.x) {
    int r = i >> 6, c = i & 63;
    ks[i] = base[(size_t)r * 1536 + 512  + hd * 64 + c];
    vs[i] = base[(size_t)r * 1536 + 1024 + hd * 64 + c];
  }
  __syncthreads();
  int wave = threadIdx.x >> 5, lane = threadIdx.x & 31;
  for (int i = wave; i < CTXL; i += 8) {
    float q0 = (float)base[(size_t)i * 1536 + hd * 64 + lane];
    float q1 = (float)base[(size_t)i * 1536 + hd * 64 + 32 + lane];
    float mx = -1e30f, sum = 0.f, o0 = 0.f, o1 = 0.f;
    for (int j = 0; j <= i; ++j) {   // causal
      float part = q0 * (float)ks[j * 64 + lane] + q1 * (float)ks[j * 64 + 32 + lane];
      part = waveRedAdd(part);
      float sc = part * 0.125f;      // 1/sqrt(64)
      float nm = fmaxf(mx, sc);
      float corr = __expf(mx - nm);
      float e = __expf(sc - nm);
      sum = sum * corr + e;
      o0 = o0 * corr + e * (float)vs[j * 64 + lane];
      o1 = o1 * corr + e * (float)vs[j * 64 + 32 + lane];
      mx = nm;
    }
    float inv = 1.f / sum;
    bf16_t* orow = out + (size_t)(b * CTXL + i) * WID + hd * 64;
    orow[lane]      = (bf16_t)(o0 * inv);
    orow[lane + 32] = (bf16_t)(o1 * inv);
  }
}

// ---------------- gather token (pos_y+LP-1) + final LN -> bf16 ----------------
__global__ __launch_bounds__(256) void gatherln_kernel(const float* __restrict__ h,
                                                       const int* __restrict__ posy,
                                                       const float* __restrict__ g,
                                                       const float* __restrict__ bt,
                                                       bf16_t* __restrict__ out) {
  int b = blockIdx.x * 8 + (threadIdx.x >> 5);
  if (b >= BSEQ) return;
  int n = b % NSEQ;
  int t = posy[n] + LPRM - 1;
  ln_row(h + ((size_t)b * CTXL + t) * WID, g, bt, out + (size_t)b * WID, threadIdx.x & 31);
}

// ---------------- row L2-normalize (1024-wide rows), wave per row ----------------
__global__ __launch_bounds__(256) void rownorm_kernel(const float* __restrict__ in,
                                                      float* __restrict__ out, int rows) {
  int r = blockIdx.x * 8 + (threadIdx.x >> 5);
  if (r >= rows) return;
  int lane = threadIdx.x & 31;
  const float* x = in + (size_t)r * EMB;
  float ss = 0.f;
  float4 xv[8];
  for (int j = 0; j < 8; ++j) {
    xv[j] = ((const float4*)x)[j * 32 + lane];
    ss += xv[j].x * xv[j].x + xv[j].y * xv[j].y + xv[j].z * xv[j].z + xv[j].w * xv[j].w;
  }
  ss = waveRedAdd(ss);
  float inv = rsqrtf(ss);
  float* o = out + (size_t)r * EMB;
  for (int j = 0; j < 8; ++j) {
    float4 v = xv[j];
    ((float4*)o)[j * 32 + lane] = make_float4(v.x * inv, v.y * inv, v.z * inv, v.w * inv);
  }
}

// ---------------- mean over N sequences per prompt ----------------
__global__ __launch_bounds__(256) void mean_kernel(const float* __restrict__ feats,
                                                   float* __restrict__ pf) {
  int p = blockIdx.x;
  int e = blockIdx.y * blockDim.x + threadIdx.x;   // 0..1023
  float s = 0.f;
  for (int n = 0; n < NSEQ; ++n)
    s += feats[((size_t)(p * NSEQ + n)) * EMB + e];
  pf[(size_t)p * EMB + e] = s * (1.f / (float)NSEQ);
}

// =======================================================================
extern "C" void kernel_launch(void* const* d_in, const int* in_sizes, int n_in,
                              void* d_out, int out_size, void* d_ws, size_t ws_size,
                              hipStream_t stream) {
  (void)in_sizes; (void)n_in; (void)out_size; (void)ws_size;
  const int*   text    = (const int*)  d_in[0];
  const float* prompt  = (const float*)d_in[1];
  const float* tok_emb = (const float*)d_in[2];
  const float* pos_emb = (const float*)d_in[3];
  const float* qkv_w   = (const float*)d_in[4];
  const float* qkv_b   = (const float*)d_in[5];
  const float* out_w   = (const float*)d_in[6];
  const float* out_b   = (const float*)d_in[7];
  const float* ln1_w   = (const float*)d_in[8];
  const float* ln1_b   = (const float*)d_in[9];
  const float* ln2_w   = (const float*)d_in[10];
  const float* ln2_b   = (const float*)d_in[11];
  const float* fc_w    = (const float*)d_in[12];
  const float* fc_b    = (const float*)d_in[13];
  const float* proj_w  = (const float*)d_in[14];
  const float* proj_b  = (const float*)d_in[15];
  const float* lnf_w   = (const float*)d_in[16];
  const float* lnf_b   = (const float*)d_in[17];
  const float* text_pr = (const float*)d_in[18];

  char* wsb = (char*)d_ws;
  size_t off = 0;
  auto wsAlloc = [&](size_t bytes) -> void* {
    void* p = wsb + off;
    off += (bytes + 255) & ~(size_t)255;
    return p;
  };
  float*  h    = (float*) wsAlloc((size_t)NTOK * WID * 4);
  bf16_t* ybf  = (bf16_t*)wsAlloc((size_t)NTOK * WID * 2);
  bf16_t* big  = (bf16_t*)wsAlloc((size_t)NTOK * 4 * WID * 2);   // qkv (1536) / fc (2048)
  bf16_t* qkvW = (bf16_t*)wsAlloc((size_t)NLAY * 3 * WID * WID * 2);
  bf16_t* outW = (bf16_t*)wsAlloc((size_t)NLAY * WID * WID * 2);
  bf16_t* fcW  = (bf16_t*)wsAlloc((size_t)NLAY * 4 * WID * WID * 2);
  bf16_t* prW  = (bf16_t*)wsAlloc((size_t)NLAY * WID * 4 * WID * 2);
  bf16_t* tpT  = (bf16_t*)wsAlloc((size_t)WID * EMB * 2);
  int*    posy = (int*)   wsAlloc(NSEQ * 4);
  bf16_t* gbuf = (bf16_t*)wsAlloc((size_t)BSEQ * WID * 2);
  float*  feats= (float*) wsAlloc((size_t)BSEQ * EMB * 4);
  float*  pf   = (float*) wsAlloc((size_t)NP * EMB * 4);

  auto cvt = [&](const float* s, bf16_t* d, size_t n) {
    f2bf_kernel<<<dim3((unsigned)((n + 255) / 256)), 256, 0, stream>>>(s, d, n);
  };
  cvt(qkv_w, qkvW, (size_t)NLAY * 3 * WID * WID);
  cvt(out_w, outW, (size_t)NLAY * WID * WID);
  cvt(fc_w,  fcW,  (size_t)NLAY * 4 * WID * WID);
  cvt(proj_w, prW, (size_t)NLAY * WID * 4 * WID);
  f2bf_t_kernel<<<dim3((WID * EMB + 255) / 256), 256, 0, stream>>>(text_pr, tpT);

  posy_kernel<<<dim3(1), 128, 0, stream>>>(text, posy);
  embed_kernel<<<dim3(NTOK / 8), 256, 0, stream>>>(text, prompt, tok_emb, pos_emb, posy, h);

  auto gGrid = [](int M, int N) {
    return dim3((unsigned)(((size_t)(M / 32) * (N / 64) + 7) / 8));
  };

  for (int l = 0; l < NLAY; ++l) {
    const bf16_t* Wq = qkvW + (size_t)l * 3 * WID * WID;
    const bf16_t* Wo = outW + (size_t)l * WID * WID;
    const bf16_t* Wf = fcW  + (size_t)l * 4 * WID * WID;
    const bf16_t* Wp = prW  + (size_t)l * WID * 4 * WID;

    // y = LN1(h) -> bf16
    ln_kernel<<<dim3(NTOK / 8), 256, 0, stream>>>(h, ln1_w + l * WID, ln1_b + l * WID, ybf, NTOK);
    // qkv = y @ Wq^T + bq  (bf16 out)
    gemm_nt_wmma<0><<<gGrid(NTOK, 3 * WID), 256, 0, stream>>>(
        ybf, Wq, qkv_b + (size_t)l * 3 * WID, big, NTOK, 3 * WID, WID);
    // attention -> ybf (bf16)
    attn_kernel<<<dim3(BSEQ * 8), 256, 0, stream>>>(big, ybf);
    // h += attn_out @ Wo^T + bo
    gemm_nt_wmma<2><<<gGrid(NTOK, WID), 256, 0, stream>>>(
        ybf, Wo, out_b + (size_t)l * WID, h, NTOK, WID, WID);
    // y = LN2(h) -> bf16
    ln_kernel<<<dim3(NTOK / 8), 256, 0, stream>>>(h, ln2_w + l * WID, ln2_b + l * WID, ybf, NTOK);
    // u = gelu(y @ Wf^T + bf) -> bf16
    gemm_nt_wmma<1><<<gGrid(NTOK, 4 * WID), 256, 0, stream>>>(
        ybf, Wf, fc_b + (size_t)l * 4 * WID, big, NTOK, 4 * WID, WID);
    // h += u @ Wp^T + bp
    gemm_nt_wmma<2><<<gGrid(NTOK, WID), 256, 0, stream>>>(
        big, Wp, proj_b + (size_t)l * WID, h, NTOK, WID, 4 * WID);
  }

  // final LN at gathered token -> gbuf (1024 x 512 bf16)
  gatherln_kernel<<<dim3(BSEQ / 8), 256, 0, stream>>>(h, posy, lnf_w, lnf_b, gbuf);
  // feats = gbuf @ tpT^T  (1024 x 1024 fp32)
  gemm_nt_wmma<3><<<gGrid(BSEQ, EMB), 256, 0, stream>>>(
      gbuf, tpT, (const float*)nullptr, feats, BSEQ, EMB, WID);
  // per-row normalize, mean over N, normalize -> d_out
  rownorm_kernel<<<dim3(BSEQ / 8), 256, 0, stream>>>(feats, feats, BSEQ);
  mean_kernel<<<dim3(NP, EMB / 256), 256, 0, stream>>>(feats, pf);
  rownorm_kernel<<<dim3(1), 256, 0, stream>>>(pf, (float*)d_out, NP);
}
